// KANLayer_90005334655801
// MI455X (gfx1250) — matrix operand order repaired and checked by reading, not compiled, not run on hardware
//
#include <hip/hip_runtime.h>
#include <hip/hip_bf16.h>

// ---------------- problem constants ----------------
#define B_DIM   4096
#define IN_DIM  2048
#define OUT_DIM 2048
#define GRID_SZ 5
#define ORDER   3
#define NCOEFF  (GRID_SZ + ORDER)         // 8
#define NKNOT   (GRID_SZ + 2 * ORDER + 1) // 12
#define KAUG    9                          // 1 (base) + 8 (spline)
#define K_TOT   (IN_DIM * KAUG)            // 18432 = 576 * 32
#define QSCALE  32.0f                      // Q_LEVELS / 2

#define KSTAGE  32                         // K depth per LDS stage (one WMMA K)
#define NSTAGE  (K_TOT / KSTAGE)           // 576
#define MT      128                        // block tile M
#define NT      128                        // block tile N

typedef __attribute__((ext_vector_type(16))) __bf16 v16bf;
typedef __attribute__((ext_vector_type(8)))  __bf16 v8bf;
typedef __attribute__((ext_vector_type(8)))  float  v8f;
typedef __attribute__((ext_vector_type(4)))  unsigned int u32x4;
typedef __attribute__((ext_vector_type(8)))  int i32x8;
typedef __attribute__((ext_vector_type(4)))  int i32x4;

// ---------------------------------------------------------------------------
// Pass 1: augmented activation matrix A[b][i*9 + c] (bf16)
// ---------------------------------------------------------------------------
__global__ void __launch_bounds__(256)
kan_build_A(const float* __restrict__ x, const float* __restrict__ grid,
            __bf16* __restrict__ A)
{
    long t = (long)blockIdx.x * 256 + threadIdx.x;     // over B*IN
    if (t >= (long)B_DIM * IN_DIM) return;
    int  i = (int)(t % IN_DIM);
    float xv = x[t];

    float g[NKNOT];
    const float* gp = grid + (long)i * NKNOT;
#pragma unroll
    for (int j = 0; j < NKNOT; ++j) g[j] = gp[j];

    float bas[NKNOT - 1];
#pragma unroll
    for (int j = 0; j < NKNOT - 1; ++j)
        bas[j] = (xv >= g[j] && xv < g[j + 1]) ? 1.0f : 0.0f;

#pragma unroll
    for (int p = 1; p <= ORDER; ++p) {
#pragma unroll
        for (int j = 0; j < NKNOT - 1 - p; ++j) {
            float t1 = (xv - g[j])         / (g[j + p]     - g[j]     + 1e-8f) * bas[j];
            float t2 = (g[j + p + 1] - xv) / (g[j + p + 1] - g[j + 1] + 1e-8f) * bas[j + 1];
            bas[j] = t1 + t2;
        }
    }

    __bf16* out = A + t * KAUG;
    out[0] = (__bf16)xv;
#pragma unroll
    for (int c = 0; c < NCOEFF; ++c) out[1 + c] = (__bf16)bas[c];
}

// ---------------------------------------------------------------------------
// Pass 2: augmented quantized weight matrix Wq[o][i*9 + c] (bf16)
// ---------------------------------------------------------------------------
__global__ void __launch_bounds__(256)
kan_build_W(const float* __restrict__ bw, const float* __restrict__ sw,
            __bf16* __restrict__ Wq)
{
    long t = (long)blockIdx.x * 256 + threadIdx.x;     // over OUT*IN
    if (t >= (long)OUT_DIM * IN_DIM) return;

    __bf16* out = Wq + t * KAUG;
    out[0] = (__bf16)bw[t];
    const float* s = sw + t * NCOEFF;
#pragma unroll
    for (int c = 0; c < NCOEFF; ++c) {
        float q = rintf(s[c] * QSCALE) * (1.0f / QSCALE);
        out[1 + c] = (__bf16)q;
    }
}

// ---------------------------------------------------------------------------
// TDM: DMA a 2D tile (KSTAGE cols x 128 rows, bf16, row stride K_TOT) from
// global to LDS. D# layout per CDNA5 ISA:
//  group0: [1:0]=count=1, [63:32]=lds_addr, [120:64]=global_addr, [127:126]=2
//  group1: [17:16]=data_size(1 -> 2B), [79:48]=tensor_dim0, [111:80]=tensor_dim1,
//          [127:112]=tile_dim0, [143:128]=tile_dim1, [207:160]=tensor_dim0_stride
// Tiles are always fully in-bounds here, so tensor_dims are just "large".
// Toolchain is the 6-arg (clang-23 / therock) builtin variant.
// ---------------------------------------------------------------------------
__device__ __forceinline__ void tdm_load_tile(unsigned lds_off, const __bf16* gptr)
{
    unsigned long long ga = (unsigned long long)(uintptr_t)gptr;

    u32x4 g0;
    g0[0] = 1u;                                          // count=1, user mode
    g0[1] = lds_off;                                     // lds_addr (bytes)
    g0[2] = (unsigned)(ga & 0xFFFFFFFFu);                // global_addr[31:0]
    g0[3] = (unsigned)((ga >> 32) & 0x01FFFFFFu) | (2u << 30);  // addr[56:32] | type=2

    const unsigned TD0 = 1u << 20;                       // tensor_dim0 (no OOB)
    const unsigned TD1 = 1u << 20;                       // tensor_dim1 (no OOB)
    i32x8 g1;
    g1[0] = (int)(1u << 16);                             // data_size = 1 (2 bytes)
    g1[1] = (int)((TD0 & 0xFFFFu) << 16);                // tensor_dim0[15:0]
    g1[2] = (int)((TD0 >> 16) | ((TD1 & 0xFFFFu) << 16));// dim0[31:16] | dim1[15:0]
    g1[3] = (int)((TD1 >> 16) | ((unsigned)KSTAGE << 16));// dim1[31:16] | tile_dim0
    g1[4] = MT;                                          // tile_dim1=128, tile_dim2=0
    g1[5] = (int)K_TOT;                                  // tensor_dim0_stride[31:0]
    g1[6] = 0;                                           // stride[47:32]=0, dim1_stride lo=0
    g1[7] = 0;

    i32x4 z4 = {0, 0, 0, 0};
    i32x8 z8 = {0, 0, 0, 0, 0, 0, 0, 0};
    __builtin_amdgcn_tensor_load_to_lds(g0, g1, z4, z4, z8, 0);
}

// ---------------------------------------------------------------------------
// Pass 3: C = A(4096xK) * Wq(2048xK)^T with bf16 WMMA.
// Block = 256 threads (8 wave32s, 4(M) x 2(N)); block tile 128x128.
// Wave 0 drives the TDM double-buffer (TENSORcnt); all waves read fragments
// from LDS (ds_load_b128) and issue 8 v_wmma_f32_16x16x32_bf16 per stage.
// LDS: 2 buffers x (A 128x32 + B 128x32) bf16 = 32 KB.
// ---------------------------------------------------------------------------
__global__ void __launch_bounds__(256)
kan_gemm(const __bf16* __restrict__ A, const __bf16* __restrict__ W,
         float* __restrict__ C)
{
    __shared__ alignas(16) __bf16 As[2][MT * KSTAGE];
    __shared__ alignas(16) __bf16 Bs[2][NT * KSTAGE];

    const int lane = threadIdx.x & 31;
    const int wid  = threadIdx.x >> 5;
    const int wm   = wid & 3;            // 0..3 -> M (32 rows each)
    const int wn   = wid >> 2;           // 0..1 -> N (64 cols each)
    const int l15  = lane & 15;
    const int lhi  = lane >> 4;          // 0 or 1

    const long mBlk = (long)blockIdx.y * MT;
    const long nBlk = (long)blockIdx.x * NT;
    const __bf16* gA = A + mBlk * (long)K_TOT;   // block's A tile rows
    const __bf16* gB = W + nBlk * (long)K_TOT;   // block's B tile rows

    // per-lane intra-buffer element offsets (ISA 7.12.2 16-bit layouts)
    //  A 16x32: lane m holds K {0..7,16..23}; lane m+16 holds {8..15,24..31}
    //  B 32x16: lane n holds K {0..7,16..23}; lane n+16 holds {8..15,24..31}
    const int aOff0 = (wm * 32      + l15) * KSTAGE + lhi * 8;
    const int aOff1 = (wm * 32 + 16 + l15) * KSTAGE + lhi * 8;
    int bOff[4];
#pragma unroll
    for (int j = 0; j < 4; ++j)
        bOff[j] = (wn * 64 + j * 16 + l15) * KSTAGE + lhi * 8;

    v8f acc[2][4] = {};

    if (wid == 0) {   // prologue: stage 0 DMA
        tdm_load_tile((unsigned)(uintptr_t)&As[0][0], gA);
        tdm_load_tile((unsigned)(uintptr_t)&Bs[0][0], gB);
    }

    for (int s = 0; s < NSTAGE; ++s) {
        const int buf = s & 1;
        __syncthreads();                 // everyone done reading buf^1
        if (wid == 0) {
            if (s + 1 < NSTAGE) {
                tdm_load_tile((unsigned)(uintptr_t)&As[buf ^ 1][0], gA + (long)(s + 1) * KSTAGE);
                tdm_load_tile((unsigned)(uintptr_t)&Bs[buf ^ 1][0], gB + (long)(s + 1) * KSTAGE);
                __builtin_amdgcn_s_wait_tensorcnt((short)2);  // stage s complete
            } else {
                __builtin_amdgcn_s_wait_tensorcnt((short)0);
            }
        }
        __syncthreads();                 // buf is ready for all waves

        // fragment loads from LDS (ds_load_b128 pairs)
        const __bf16* ap0 = &As[buf][aOff0];
        const __bf16* ap1 = &As[buf][aOff1];
        v16bf a0, a1, b[4];
        {
            v8bf lo, hi;
            lo = *(const v8bf*)(ap0);      hi = *(const v8bf*)(ap0 + 16);
#pragma unroll
            for (int e = 0; e < 8; ++e) { a0[e] = lo[e]; a0[e + 8] = hi[e]; }
            lo = *(const v8bf*)(ap1);      hi = *(const v8bf*)(ap1 + 16);
#pragma unroll
            for (int e = 0; e < 8; ++e) { a1[e] = lo[e]; a1[e + 8] = hi[e]; }
#pragma unroll
            for (int j = 0; j < 4; ++j) {
                const __bf16* bp = &Bs[buf][bOff[j]];
                lo = *(const v8bf*)(bp);   hi = *(const v8bf*)(bp + 16);
#pragma unroll
                for (int e = 0; e < 8; ++e) { b[j][e] = lo[e]; b[j][e + 8] = hi[e]; }
            }
        }

#pragma unroll
        for (int j = 0; j < 4; ++j) {
            acc[0][j] = __builtin_amdgcn_wmma_f32_16x16x32_bf16(false, a0, false, b[j], (short)0, acc[0][j], false, false);
            acc[1][j] = __builtin_amdgcn_wmma_f32_16x16x32_bf16(false, a1, false, b[j], (short)0, acc[1][j], false, false);
        }
    }

    // C/D layout: lane n = lane&15; VGPR r -> M = r + (lane>=16 ? 8 : 0)
    const long m0 = mBlk + wm * 32;
    const long n0 = nBlk + wn * 64;
#pragma unroll
    for (int mi = 0; mi < 2; ++mi) {
#pragma unroll
        for (int ni = 0; ni < 4; ++ni) {
            long mBase = m0 + mi * 16 + lhi * 8;
            long nIdx  = n0 + ni * 16 + l15;
#pragma unroll
            for (int r = 0; r < 8; ++r)
                C[(mBase + r) * (long)OUT_DIM + nIdx] = acc[mi][ni][r];
        }
    }
}

// ---------------------------------------------------------------------------
extern "C" void kernel_launch(void* const* d_in, const int* in_sizes, int n_in,
                              void* d_out, int out_size, void* d_ws, size_t ws_size,
                              hipStream_t stream)
{
    const float* x    = (const float*)d_in[0];   // (B, IN)
    const float* bw   = (const float*)d_in[1];   // (OUT, IN)
    const float* sw   = (const float*)d_in[2];   // (OUT, IN, 8)
    const float* grid = (const float*)d_in[3];   // (IN, 12)
    float* out = (float*)d_out;                  // (B, OUT)

    const size_t A_BYTES = (size_t)B_DIM * K_TOT * sizeof(__bf16);   // ~151 MB
    __bf16* Apack = (__bf16*)d_ws;
    __bf16* Wpack = (__bf16*)((char*)d_ws + A_BYTES);                // ~75.5 MB

    {
        long n = (long)B_DIM * IN_DIM;
        kan_build_A<<<dim3((unsigned)((n + 255) / 256)), dim3(256), 0, stream>>>(x, grid, Apack);
    }
    {
        long n = (long)OUT_DIM * IN_DIM;
        kan_build_W<<<dim3((unsigned)((n + 255) / 256)), dim3(256), 0, stream>>>(bw, sw, Wpack);
    }
    {
        dim3 g(OUT_DIM / NT, B_DIM / MT);   // (16, 32)
        kan_gemm<<<g, dim3(256), 0, stream>>>(Apack, Wpack, out);
    }
}
